// ConvMultiStepAttention_50027779064419
// MI455X (gfx1250) — matrix operand order repaired and checked by reading, not compiled, not run on hardware
//
#include <hip/hip_runtime.h>
#include <hip/hip_bf16.h>

typedef __attribute__((ext_vector_type(16))) __bf16        v16bf;
typedef __attribute__((ext_vector_type(8)))  float         v8f;
typedef __attribute__((ext_vector_type(8)))  unsigned short v8us;
typedef __attribute__((ext_vector_type(16))) unsigned short v16us;

#define B_ 8
#define C_ 512
#define T_ 2048
#define S_ 2048
#define TM 128
#define TN 128
#define TK 32
#define LDSS 40   /* padded LDS row stride (elems): 80B, 16B-aligned, bank-spread */
#define SCALE_W 0.70710678118654752440f

static __device__ __forceinline__ unsigned short f32_to_bf16(float f) {
  unsigned int u = __builtin_bit_cast(unsigned int, f);
  u += 0x7fffu + ((u >> 16) & 1u);            // round-to-nearest-even
  return (unsigned short)(u >> 16);
}

// A fragment: 16x32 bf16, M rows striped in lanes: lane<16 -> row=lane, K=0..7 & 16..23;
// lane>=16 -> row=lane-16, K=8..15 & 24..31. LDS layout [m][k], two 16B runs per lane.
static __device__ __forceinline__ v16bf frag_a(const unsigned short* s, int m0, int lane) {
  int r = lane & 15, hi = lane >> 4;
  const unsigned short* p = s + (m0 + r) * LDSS + hi * 8;
  v8us lo = *(const v8us*)p;
  v8us hv = *(const v8us*)(p + 16);
  v16us c = __builtin_shufflevector(lo, hv, 0,1,2,3,4,5,6,7,8,9,10,11,12,13,14,15);
  return __builtin_bit_cast(v16bf, c);
}

// B fragment: 32x16 bf16, N cols in lanes: lane<16 -> col=lane, K=0..7 & 16..23; lane>=16 -> K=8..15 & 24..31.
// LDS staged transposed [n][k], two 16B runs per lane.
static __device__ __forceinline__ v16bf frag_b(const unsigned short* s, int n0, int lane) {
  int r = lane & 15, hi = lane >> 4;
  const unsigned short* p = s + (n0 + r) * LDSS + hi * 8;
  v8us lo = *(const v8us*)p;
  v8us hv = *(const v8us*)(p + 16);
  v16us c = __builtin_shufflevector(lo, hv, 0,1,2,3,4,5,6,7,8,9,10,11,12,13,14,15);
  return __builtin_bit_cast(v16bf, c);
}

// CDNA5 async global->LDS copy (ASYNCcnt path), 16B per lane per issue.
// vdst operand carries the LDS byte address, vaddr the 64-bit global address.
static __device__ __forceinline__ void async_b128(unsigned lds_off, unsigned long long gaddr) {
  asm volatile("global_load_async_to_lds_b128 %0, %1, off"
               :: "v"(lds_off), "v"(gaddr) : "memory");
}
static __device__ __forceinline__ void wait_async0() {
  asm volatile("s_wait_asynccnt 0x0" ::: "memory");
}

// ---------------------------------------------------------------- cvt f32->bf16
__global__ __launch_bounds__(256)
void cvt_bf16_kernel(const float* __restrict__ src, unsigned short* __restrict__ dst, int n) {
  int i = blockIdx.x * blockDim.x + threadIdx.x;
  int stride = gridDim.x * blockDim.x;
  for (; i < n; i += stride) dst[i] = f32_to_bf16(src[i]);
}

// ------------------------------------------- target = (base + W@x + b) * sqrt(.5)
__global__ __launch_bounds__(256)
void linear_bf16_wmma_kernel(const float* __restrict__ x,     // [B,C,T]
                             const float* __restrict__ base,  // [B,C,T]
                             const float* __restrict__ W,     // [C,C]
                             const float* __restrict__ bias,  // [C]
                             unsigned short* __restrict__ tgt) // [B,C,T] bf16
{
  __shared__ unsigned short sA[TM * LDSS];
  __shared__ unsigned short sB[TN * LDSS];
  const int ntile = T_ / TN, mtile = C_ / TM;
  int bid = blockIdx.x;
  int b   = bid / (ntile * mtile);
  int rem = bid % (ntile * mtile);
  int m0  = (rem / ntile) * TM;  // output-channel base
  int n0  = (rem % ntile) * TN;  // t base
  int tid = threadIdx.x, lane = tid & 31, wave = tid >> 5;
  int wm = (wave & 3) * 32, wn = (wave >> 2) * 64;
  const float* xb = x + (size_t)b * C_ * T_;
  v8f acc[2][4] = {};
  for (int k0 = 0; k0 < C_; k0 += TK) {
    __syncthreads();
    { // A tile: W[m][k], row-major, direct copy with f32->bf16 convert
      int m = tid >> 1, h = (tid & 1) * 16;
      const float* src = W + (size_t)(m0 + m) * C_ + k0 + h;
      unsigned short* dst = sA + m * LDSS + h;
#pragma unroll
      for (int i = 0; i < 16; ++i) dst[i] = f32_to_bf16(src[i]);
    }
    { // B tile: x[k][n] -> LDS [n][k] transpose with convert
      int kk = tid >> 3, seg = (tid & 7) * 16;
      const float* src = xb + (size_t)(k0 + kk) * T_ + n0 + seg;
#pragma unroll
      for (int i = 0; i < 16; ++i) sB[(seg + i) * LDSS + kk] = f32_to_bf16(src[i]);
    }
    __syncthreads();
    v16bf bfr[4];
#pragma unroll
    for (int fn = 0; fn < 4; ++fn) bfr[fn] = frag_b(sB, wn + fn * 16, lane);
#pragma unroll
    for (int fm = 0; fm < 2; ++fm) {
      v16bf afr = frag_a(sA, wm + fm * 16, lane);
#pragma unroll
      for (int fn = 0; fn < 4; ++fn)
        acc[fm][fn] = __builtin_amdgcn_wmma_f32_16x16x32_bf16(
            false, afr, false, bfr[fn], (short)0, acc[fm][fn], false, false);
    }
  }
  int r = lane & 15, hi = lane >> 4;
  size_t bCT = (size_t)b * C_ * T_;
#pragma unroll
  for (int fm = 0; fm < 2; ++fm)
#pragma unroll
    for (int fn = 0; fn < 4; ++fn)
#pragma unroll
      for (int i = 0; i < 8; ++i) {
        int o = m0 + wm + fm * 16 + i + hi * 8;   // C/D layout: VGPR i -> M=i (+8 for hi lanes)
        int t = n0 + wn + fn * 16 + r;            // N = lane&15
        size_t off = bCT + (size_t)o * T_ + t;
        float v = (base[off] + acc[fm][fn][i] + bias[o]) * SCALE_W;
        tgt[off] = f32_to_bf16(v);
      }
}

// ------------------------------------------- scores[b,t,s] = sum_c tgt[c,t]*encT[c,s]
__global__ __launch_bounds__(256)
void scores_wmma_kernel(const unsigned short* __restrict__ tgt,   // [B,C,T] bf16
                        const unsigned short* __restrict__ encT,  // [B,C,S] bf16
                        float* __restrict__ attn)                 // [B,T,S] raw f32 scores
{
  __shared__ unsigned short sA[TM * LDSS];
  __shared__ unsigned short sB[TN * LDSS];
  const int ntile = S_ / TN, mtile = T_ / TM;
  int bid = blockIdx.x;
  int b   = bid / (ntile * mtile);
  int rem = bid % (ntile * mtile);
  int m0  = (rem / ntile) * TM;  // t base
  int n0  = (rem % ntile) * TN;  // s base
  int tid = threadIdx.x, lane = tid & 31, wave = tid >> 5;
  int wm = (wave & 3) * 32, wn = (wave >> 2) * 64;
  const unsigned short* tb = tgt + (size_t)b * C_ * T_;
  const unsigned short* eb = encT + (size_t)b * C_ * S_;
  v8f acc[2][4] = {};
  int kk = tid >> 3, seg = (tid & 7) * 16;
  for (int k0 = 0; k0 < C_; k0 += TK) {
    __syncthreads();
    { // A: tgt[k=c][m=t] (t contiguous) -> LDS [m][k]
      const unsigned short* src = tb + (size_t)(k0 + kk) * T_ + m0 + seg;
      if (k0 + TK < C_) __builtin_prefetch(src + (size_t)TK * T_, 0, 1);
      v8us u0 = ((const v8us*)src)[0];
      v8us u1 = ((const v8us*)src)[1];
#pragma unroll
      for (int i = 0; i < 8; ++i) {
        sA[(seg + i) * LDSS + kk]     = u0[i];
        sA[(seg + 8 + i) * LDSS + kk] = u1[i];
      }
    }
    { // B: encT[k=c][n=s] (s contiguous) -> LDS [n][k]
      const unsigned short* src = eb + (size_t)(k0 + kk) * S_ + n0 + seg;
      if (k0 + TK < C_) __builtin_prefetch(src + (size_t)TK * S_, 0, 1);
      v8us u0 = ((const v8us*)src)[0];
      v8us u1 = ((const v8us*)src)[1];
#pragma unroll
      for (int i = 0; i < 8; ++i) {
        sB[(seg + i) * LDSS + kk]     = u0[i];
        sB[(seg + 8 + i) * LDSS + kk] = u1[i];
      }
    }
    __syncthreads();
    v16bf bfr[4];
#pragma unroll
    for (int fn = 0; fn < 4; ++fn) bfr[fn] = frag_b(sB, wn + fn * 16, lane);
#pragma unroll
    for (int fm = 0; fm < 2; ++fm) {
      v16bf afr = frag_a(sA, wm + fm * 16, lane);
#pragma unroll
      for (int fn = 0; fn < 4; ++fn)
        acc[fm][fn] = __builtin_amdgcn_wmma_f32_16x16x32_bf16(
            false, afr, false, bfr[fn], (short)0, acc[fm][fn], false, false);
    }
  }
  int r = lane & 15, hi = lane >> 4;
  size_t bTS = (size_t)b * T_ * S_;
#pragma unroll
  for (int fm = 0; fm < 2; ++fm)
#pragma unroll
    for (int fn = 0; fn < 4; ++fn)
#pragma unroll
      for (int i = 0; i < 8; ++i) {
        int t = m0 + wm + fm * 16 + i + hi * 8;
        int s = n0 + wn + fn * 16 + r;
        attn[bTS + (size_t)t * S_ + s] = acc[fm][fn][i];
      }
}

// ------------------------------------------- row softmax over S (one wave32 per row)
__global__ __launch_bounds__(256)
void softmax_kernel(float* __restrict__ attn, unsigned short* __restrict__ attn_bf) {
  int lane = threadIdx.x & 31, wave = threadIdx.x >> 5;
  size_t row = (size_t)blockIdx.x * 8 + wave;   // < B*T
  float* p = attn + row * (size_t)S_;
  unsigned short* q = attn_bf + row * (size_t)S_;
  const float4* p4 = (const float4*)p;
  float4 v[16];
  float m = -3.402823466e38f;
#pragma unroll
  for (int i = 0; i < 16; ++i) {
    v[i] = p4[i * 32 + lane];
    m = fmaxf(m, fmaxf(fmaxf(v[i].x, v[i].y), fmaxf(v[i].z, v[i].w)));
  }
#pragma unroll
  for (int off = 16; off > 0; off >>= 1) m = fmaxf(m, __shfl_xor(m, off, 32));
  float s = 0.f;
#pragma unroll
  for (int i = 0; i < 16; ++i) {
    v[i].x = __expf(v[i].x - m);
    v[i].y = __expf(v[i].y - m);
    v[i].z = __expf(v[i].z - m);
    v[i].w = __expf(v[i].w - m);
    s += v[i].x + v[i].y + v[i].z + v[i].w;
  }
#pragma unroll
  for (int off = 16; off > 0; off >>= 1) s += __shfl_xor(s, off, 32);
  float inv = 1.0f / s;
  float4* o4 = (float4*)p;
#pragma unroll
  for (int i = 0; i < 16; ++i) {
    v[i].x *= inv; v[i].y *= inv; v[i].z *= inv; v[i].w *= inv;
    o4[i * 32 + lane] = v[i];
    int e = (i * 32 + lane) * 4;
    q[e + 0] = f32_to_bf16(v[i].x);
    q[e + 1] = f32_to_bf16(v[i].y);
    q[e + 2] = f32_to_bf16(v[i].z);
    q[e + 3] = f32_to_bf16(v[i].w);
  }
}

// ------------------------------------------- ctx[b,c,t] = sum_s attn[t,s]*encC[c,s]
// Double-buffered: async global->LDS copies (ASYNCcnt) overlap next tile with WMMA.
__global__ __launch_bounds__(256)
void ctx_wmma_kernel(const unsigned short* __restrict__ attn_bf, // [B,T,S] bf16
                     const unsigned short* __restrict__ encC,    // [B,C,S] bf16
                     float* __restrict__ ctx)                    // [B,C,T] f32
{
  __shared__ unsigned short sA[2][TM * LDSS];
  __shared__ unsigned short sB[2][TN * LDSS];
  const int ntile = C_ / TN, mtile = T_ / TM;
  int bid = blockIdx.x;
  int b   = bid / (ntile * mtile);
  int rem = bid % (ntile * mtile);
  int m0  = (rem / ntile) * TM;  // t base
  int n0  = (rem % ntile) * TN;  // c base
  int tid = threadIdx.x, lane = tid & 31, wave = tid >> 5;
  int wm = (wave & 3) * 32, wn = (wave >> 2) * 64;
  const unsigned short* ab = attn_bf + (size_t)b * T_ * S_;
  const unsigned short* eb = encC + (size_t)b * C_ * S_;
  v8f acc[2][4] = {};
  int mrow = tid >> 1, h = (tid & 1) * 16;   // 16-elem (32B) run per thread per tile row
  const unsigned short* srcA0 = ab + (size_t)(m0 + mrow) * S_ + h;   // [m=t][k=s], k contiguous
  const unsigned short* srcB0 = eb + (size_t)(n0 + mrow) * S_ + h;   // [n=c][k=s], k contiguous
  unsigned ldsA[2], ldsB[2];
  ldsA[0] = (unsigned)(size_t)&sA[0][mrow * LDSS + h];
  ldsA[1] = (unsigned)(size_t)&sA[1][mrow * LDSS + h];
  ldsB[0] = (unsigned)(size_t)&sB[0][mrow * LDSS + h];
  ldsB[1] = (unsigned)(size_t)&sB[1][mrow * LDSS + h];

  // prologue: async-stage tile 0 into buffer 0 (4 x 16B per thread)
  async_b128(ldsA[0],      (unsigned long long)srcA0);
  async_b128(ldsA[0] + 16, (unsigned long long)(srcA0 + 8));
  async_b128(ldsB[0],      (unsigned long long)srcB0);
  async_b128(ldsB[0] + 16, (unsigned long long)(srcB0 + 8));

  const int NT = S_ / TK;   // 64 K-tiles
  for (int kt = 0; kt < NT; ++kt) {
    int cur = kt & 1;
    wait_async0();      // this wave's pending tile has landed in LDS
    __syncthreads();    // all waves' tiles landed; prior reads of 'nxt' buffer done
    if (kt + 1 < NT) {  // async-prefetch next tile into the other buffer
      int nk = (kt + 1) * TK;
      const unsigned short* a  = srcA0 + nk;
      const unsigned short* bb = srcB0 + nk;
      int nxt = cur ^ 1;
      async_b128(ldsA[nxt],      (unsigned long long)a);
      async_b128(ldsA[nxt] + 16, (unsigned long long)(a + 8));
      async_b128(ldsB[nxt],      (unsigned long long)bb);
      async_b128(ldsB[nxt] + 16, (unsigned long long)(bb + 8));
    }
    v16bf bfr[4];
#pragma unroll
    for (int fn = 0; fn < 4; ++fn) bfr[fn] = frag_b(sB[cur], wn + fn * 16, lane);
#pragma unroll
    for (int fm = 0; fm < 2; ++fm) {
      v16bf afr = frag_a(sA[cur], wm + fm * 16, lane);
#pragma unroll
      for (int fn = 0; fn < 4; ++fn)
        acc[fm][fn] = __builtin_amdgcn_wmma_f32_16x16x32_bf16(
            false, afr, false, bfr[fn], (short)0, acc[fm][fn], false, false);
    }
  }
  int r = lane & 15, hi = lane >> 4;
  size_t bCT = (size_t)b * C_ * T_;
#pragma unroll
  for (int fm = 0; fm < 2; ++fm)
#pragma unroll
    for (int fn = 0; fn < 4; ++fn)
#pragma unroll
      for (int i = 0; i < 8; ++i) {
        int t = m0 + wm + fm * 16 + i + hi * 8;
        int c = n0 + wn + fn * 16 + r;
        ctx[bCT + (size_t)c * T_ + t] = acc[fm][fn][i];   // transpose to [B,C,T]
      }
}

extern "C" void kernel_launch(void* const* d_in, const int* in_sizes, int n_in,
                              void* d_out, int out_size, void* d_ws, size_t ws_size,
                              hipStream_t stream) {
  (void)in_sizes; (void)n_in; (void)out_size; (void)ws_size;
  const float* base = (const float*)d_in[0];   // [B,C,T,1]
  const float* inp  = (const float*)d_in[1];   // [B,C,T,1]
  const float* encT = (const float*)d_in[2];   // [B,C,S]
  const float* encC = (const float*)d_in[3];   // [B,C,S]
  const float* W    = (const float*)d_in[4];   // [C,C]
  const float* bias = (const float*)d_in[5];   // [C]

  const size_t BCT = (size_t)B_ * C_ * T_;     // 8,388,608
  const size_t BCS = (size_t)B_ * C_ * S_;     // 8,388,608
  const size_t BTS = (size_t)B_ * T_ * S_;     // 33,554,432

  float* ctx  = (float*)d_out;                 // output 0: [B,C,T,1]
  float* attn = ctx + BCT;                     // output 1: [B,T,S]

  // workspace: bf16 shadows (~117 MB total)
  unsigned short* ws      = (unsigned short*)d_ws;
  unsigned short* tgt_bf  = ws;                // BCT elems
  unsigned short* encT_bf = tgt_bf + BCT;      // BCS elems
  unsigned short* encC_bf = encT_bf + BCS;     // BCS elems
  unsigned short* attn_bf = encC_bf + BCS;     // BTS elems

  cvt_bf16_kernel<<<4096, 256, 0, stream>>>(encT, encT_bf, (int)BCS);
  cvt_bf16_kernel<<<4096, 256, 0, stream>>>(encC, encC_bf, (int)BCS);
  linear_bf16_wmma_kernel<<<B_ * (C_ / TM) * (T_ / TN), 256, 0, stream>>>(
      inp, base, W, bias, tgt_bf);
  scores_wmma_kernel<<<B_ * (T_ / TM) * (S_ / TN), 256, 0, stream>>>(
      tgt_bf, encT_bf, attn);
  softmax_kernel<<<(B_ * T_) / 8, 256, 0, stream>>>(attn, attn_bf);
  ctx_wmma_kernel<<<B_ * (T_ / TM) * (C_ / TN), 256, 0, stream>>>(
      attn_bf, encC_bf, ctx);
}
